// HybridAttentionBlock_42348377538918
// MI455X (gfx1250) — compile-verified, matrix operand order checked by reading, not run on hardware
//
#include <hip/hip_runtime.h>

// ---------------------------------------------------------------------------
// HybridAttentionBlock for MI455X (gfx1250, wave32, WMMA + Tensor Data Mover)
//
// Pipeline (all on `stream`):
//   1. cvt_w_kernel      : Wq/Wk/Wv/Wo fp32 -> f16           (4 launches)
//   2. ln_kernel         : h = LayerNorm(x) -> f16 [4096,1024]
//   3. gemm_wmma_kernel  : Q/K/V = h @ W^T -> f16 head-major  (3 launches)
//                          (TDM double-buffered LDS tiles + WMMA)
//   4. attn_kernel       : sliding-window causal flash attention
//                          (per-wave TDM V tiles + WMMA)
//   5. gemm_wmma_kernel  : out = x + O @ Wo^T -> fp32
//
// Workspace layout (assumes ws_size >= 48 MiB):
//   [ 0MB) h16   [8MB) wq16  [10MB) wk16  [12MB) wv16  [14MB) wo16
//   [16MB) Q hm  [24MB) K hm [32MB) V hm  [40MB) O16 row-major
// ---------------------------------------------------------------------------

typedef __attribute__((ext_vector_type(16))) _Float16 v16h;
typedef __attribute__((ext_vector_type(8)))  _Float16 v8h;
typedef __attribute__((ext_vector_type(8)))  float    v8f;
typedef __attribute__((ext_vector_type(4)))  unsigned int v4u;
typedef __attribute__((ext_vector_type(8)))  int      v8i;
typedef __attribute__((ext_vector_type(4)))  int      v4i;

#define D_MODEL 1024
#define N_HEADS 16
#define D_HEAD  64
#define SEQ_L   2048
#define BATCH   2
#define BL      (BATCH * SEQ_L)   /* 4096 rows */
#define WINDOW  256

__device__ __forceinline__ v16h pack16(v8h lo, v8h hi) {
    v16h r;
#pragma unroll
    for (int i = 0; i < 8; ++i) { r[i] = lo[i]; r[i + 8] = hi[i]; }
    return r;
}

__device__ __forceinline__ v8f wmma_f16(v16h a, v16h b, v8f c) {
    return __builtin_amdgcn_wmma_f32_16x16x32_f16(false, a, false, b, (short)0, c,
                                                  false, false);
}

// ---------------------------------------------------------------------------
// Tensor Data Mover: 2-D f16 tile (tile0 elems/row x tile1 rows) from global
// to LDS, row stride = stride0 elements, optional LDS row padding.
// D# per CDNA5 ISA: group0 = {ctl, lds_addr, gaddr lo/hi},
// group1 = {flags, dims/tiles, strides}; groups 2/3 inert for 2-D tiles.
// This toolchain exposes the 6-arg builtin (extra int32x8 group, zero-filled).
// ---------------------------------------------------------------------------
__device__ __forceinline__ void tdm_load_2d(const _Float16* gsrc, unsigned lds_off,
                                            unsigned td0, unsigned td1,
                                            unsigned tile0, unsigned tile1,
                                            unsigned stride0,
                                            unsigned pad_interval,
                                            unsigned pad_amount,
                                            unsigned pad_en) {
    unsigned long long ga = (unsigned long long)(uintptr_t)gsrc;
    v4u g0;
    g0[0] = 1u;                                                  // count=1, user D#
    g0[1] = lds_off;                                             // lds_addr (bytes)
    g0[2] = (unsigned)ga;                                        // global_addr[31:0]
    g0[3] = ((unsigned)(ga >> 32) & 0x01FFFFFFu) | (2u << 30);   // addr[56:32]|type=2
    v8i g1;
    g1[0] = (int)((1u << 16) |                 // data_size = 2 bytes
                  (pad_en << 20) | (pad_interval << 22) | (pad_amount << 25));
    g1[1] = (int)((td0 & 0xFFFFu) << 16);                        // tensor_dim0 lo
    g1[2] = (int)((td0 >> 16) | ((td1 & 0xFFFFu) << 16));        // dim0 hi | dim1 lo
    g1[3] = (int)((td1 >> 16) | (tile0 << 16));                  // dim1 hi | tile_dim0
    g1[4] = (int)(tile1 & 0xFFFFu);                              // tile_dim1, tile_dim2=0
    g1[5] = (int)stride0;                                        // tensor_dim0_stride lo
    g1[6] = 0;                                                   // stride0 hi | stride1 lo
    g1[7] = 0;                                                   // stride1 hi
    v4i g2;
    g2[0] = 1; g2[1] = 0; g2[2] = 0; g2[3] = 0;                  // tensor_dim2=1, unused
    v4i g3;
    g3[0] = 0; g3[1] = 0; g3[2] = 0; g3[3] = 0;
    v8i g4;
    g4[0] = 0; g4[1] = 0; g4[2] = 0; g4[3] = 0;
    g4[4] = 0; g4[5] = 0; g4[6] = 0; g4[7] = 0;
    __builtin_amdgcn_tensor_load_to_lds(g0, g1, g2, g3, g4, 0);
}

// ---------------------------------------------------------------------------
// fp32 -> f16 weight conversion
// ---------------------------------------------------------------------------
__global__ __launch_bounds__(256) void cvt_w_kernel(const float* __restrict__ src,
                                                    _Float16* __restrict__ dst,
                                                    int n) {
    int i = (blockIdx.x * 256 + threadIdx.x) * 4;
    if (i + 3 < n) {
        float4 v = *reinterpret_cast<const float4*>(src + i);
        dst[i + 0] = (_Float16)v.x;
        dst[i + 1] = (_Float16)v.y;
        dst[i + 2] = (_Float16)v.z;
        dst[i + 3] = (_Float16)v.w;
    }
}

// ---------------------------------------------------------------------------
// LayerNorm: one 256-thread block per row of 1024
// ---------------------------------------------------------------------------
__global__ __launch_bounds__(256) void ln_kernel(const float* __restrict__ x,
                                                 const float* __restrict__ gamma,
                                                 const float* __restrict__ beta,
                                                 _Float16* __restrict__ h) {
    __shared__ float red[256];
    const int row = blockIdx.x;
    const int tid = threadIdx.x;
    const float4 v = reinterpret_cast<const float4*>(x + (size_t)row * D_MODEL)[tid];

    red[tid] = v.x + v.y + v.z + v.w;
    __syncthreads();
    for (int off = 128; off > 0; off >>= 1) {
        if (tid < off) red[tid] += red[tid + off];
        __syncthreads();
    }
    const float mu = red[0] * (1.0f / (float)D_MODEL);
    __syncthreads();

    const float dx = v.x - mu, dy = v.y - mu, dz = v.z - mu, dw = v.w - mu;
    red[tid] = dx * dx + dy * dy + dz * dz + dw * dw;
    __syncthreads();
    for (int off = 128; off > 0; off >>= 1) {
        if (tid < off) red[tid] += red[tid + off];
        __syncthreads();
    }
    const float rstd = rsqrtf(red[0] * (1.0f / (float)D_MODEL) + 1e-5f);

    const int c = tid * 4;
    _Float16* hr = h + (size_t)row * D_MODEL + c;
    hr[0] = (_Float16)(dx * rstd * gamma[c + 0] + beta[c + 0]);
    hr[1] = (_Float16)(dy * rstd * gamma[c + 1] + beta[c + 1]);
    hr[2] = (_Float16)(dz * rstd * gamma[c + 2] + beta[c + 2]);
    hr[3] = (_Float16)(dw * rstd * gamma[c + 3] + beta[c + 3]);
}

// ---------------------------------------------------------------------------
// Tiled WMMA GEMM with TDM-fed, double-buffered LDS: C[M,N] = A[M,K]*W[N,K]^T
//   mode 0: f16 head-major [B,H,L,dh] (QKV);  mode 2: fp32 + residual (Wo)
// 256 threads = 8 waves; tile 128x128; k-step 32; wave grid 2(M) x 4(N).
// LDS rows: 32 data halves + 16 pad halves (TDM pad: 16 dwords + 8 dwords)
// keep every b128 fragment read 16B-aligned.
// ---------------------------------------------------------------------------
#define GT_M 128
#define GT_N 128
#define GT_K 32
#define GT_PITCH 48

__global__ __launch_bounds__(256) void gemm_wmma_kernel(
    const _Float16* __restrict__ A, const _Float16* __restrict__ Bw,
    void* __restrict__ dst, const float* __restrict__ resid,
    int M, int N, int K, int mode) {
    __shared__ _Float16 As[2][GT_M][GT_PITCH];
    __shared__ _Float16 Bs[2][GT_N][GT_PITCH];

    const int tid = threadIdx.x;
    const unsigned lane = tid & 31u;
    const int wid = tid >> 5;
    const int waveM = wid & 1;   // 0..1
    const int waveN = wid >> 1;  // 0..3
    const int m0 = blockIdx.y * GT_M;
    const int n0 = blockIdx.x * GT_N;
    const int lm = (int)(lane & 15u);
    const int koff = (lane < 16u) ? 0 : 8;
    const int nk = K / GT_K;

    v8f zero = {};
    v8f acc[4][2];
#pragma unroll
    for (int mi = 0; mi < 4; ++mi)
#pragma unroll
        for (int ni = 0; ni < 2; ++ni) acc[mi][ni] = zero;

    // prologue: DMA tile 0 into buffer 0 (wave 0 drives the TDM)
    if (wid == 0) {
        tdm_load_2d(A + (size_t)m0 * K, (unsigned)(uintptr_t)&As[0][0][0],
                    (unsigned)K, (unsigned)M, GT_K, GT_M, (unsigned)K, 3u, 7u, 1u);
        tdm_load_2d(Bw + (size_t)n0 * K, (unsigned)(uintptr_t)&Bs[0][0][0],
                    (unsigned)K, (unsigned)N, GT_K, GT_N, (unsigned)K, 3u, 7u, 1u);
    }

    for (int t = 0; t < nk; ++t) {
        const int cur = t & 1;
        __syncthreads();  // everyone done reading buffer cur^1 (iteration t-1)
        if (wid == 0) {
            if (t + 1 < nk) {
                const int kn = (t + 1) * GT_K;
                tdm_load_2d(A + (size_t)m0 * K + kn,
                            (unsigned)(uintptr_t)&As[cur ^ 1][0][0],
                            (unsigned)K, (unsigned)M, GT_K, GT_M, (unsigned)K, 3u, 7u, 1u);
                tdm_load_2d(Bw + (size_t)n0 * K + kn,
                            (unsigned)(uintptr_t)&Bs[cur ^ 1][0][0],
                            (unsigned)K, (unsigned)N, GT_K, GT_N, (unsigned)K, 3u, 7u, 1u);
                __builtin_amdgcn_s_wait_tensorcnt(2);  // in-order: tile t landed
            } else {
                __builtin_amdgcn_s_wait_tensorcnt(0);
            }
        }
        __syncthreads();  // tile t visible to all waves

        v16h af[4];
#pragma unroll
        for (int mi = 0; mi < 4; ++mi) {
            const _Float16* s = &As[cur][waveM * 64 + mi * 16 + lm][0];
            af[mi] = pack16(*reinterpret_cast<const v8h*>(s + koff),
                            *reinterpret_cast<const v8h*>(s + koff + 16));
        }
        v16h bf[2];
#pragma unroll
        for (int ni = 0; ni < 2; ++ni) {
            const _Float16* s = &Bs[cur][waveN * 32 + ni * 16 + lm][0];
            bf[ni] = pack16(*reinterpret_cast<const v8h*>(s + koff),
                            *reinterpret_cast<const v8h*>(s + koff + 16));
        }
#pragma unroll
        for (int mi = 0; mi < 4; ++mi)
#pragma unroll
            for (int ni = 0; ni < 2; ++ni)
                acc[mi][ni] = wmma_f16(af[mi], bf[ni], acc[mi][ni]);
    }

    // epilogue: C layout = lanes 0-15 row r, lanes 16-31 row r+8
#pragma unroll
    for (int mi = 0; mi < 4; ++mi)
#pragma unroll
        for (int ni = 0; ni < 2; ++ni)
#pragma unroll
            for (int r = 0; r < 8; ++r) {
                const int mrow = m0 + waveM * 64 + mi * 16 + ((lane < 16u) ? r : r + 8);
                const int ncol = n0 + waveN * 32 + ni * 16 + lm;
                const float v = acc[mi][ni][r];
                if (mode == 0) {  // head-major f16: [B,H,L,dh]
                    const int b = mrow >> 11, l = mrow & (SEQ_L - 1);
                    const int hh = ncol >> 6, d = ncol & (D_HEAD - 1);
                    ((_Float16*)dst)[(((size_t)(b * N_HEADS + hh)) * SEQ_L + l) * D_HEAD + d] =
                        (_Float16)v;
                } else {          // fp32 + residual
                    const size_t idx = (size_t)mrow * N + ncol;
                    ((float*)dst)[idx] = resid[idx] + v;
                }
            }
}

// ---------------------------------------------------------------------------
// Sliding-window causal attention, flash style.
// 128 threads = 4 waves; each wave owns 16 queries of one (b,h); key tiles
// of 32. V tiles arrive by per-wave TDM (TENSORcnt private to the wave),
// overlapped with the S = Q K^T WMMAs and the online softmax.
// ---------------------------------------------------------------------------
#define AT_WAVES 4

__global__ __launch_bounds__(128) void attn_kernel(const _Float16* __restrict__ Qg,
                                                   const _Float16* __restrict__ Kg,
                                                   const _Float16* __restrict__ Vg,
                                                   _Float16* __restrict__ Og) {
    __shared__ _Float16 Vs[AT_WAVES][32][D_HEAD];  // per-wave V tile (TDM dest)
    __shared__ _Float16 Ps[AT_WAVES][16][32];      // per-wave P bounce (C->A layout)

    const int tid = threadIdx.x;
    const unsigned lane = tid & 31u;
    const int wid = tid >> 5;
    const int bh = blockIdx.y;  // b*16 + h
    const int b = bh >> 4, hh = bh & 15;
    const int qi0 = blockIdx.x * (AT_WAVES * 16) + wid * 16;

    const _Float16* Qp = Qg + (size_t)bh * SEQ_L * D_HEAD;
    const _Float16* Kp = Kg + (size_t)bh * SEQ_L * D_HEAD;
    const _Float16* Vp = Vg + (size_t)bh * SEQ_L * D_HEAD;

    const int lm = (int)(lane & 15u);
    const int koff = (lane < 16u) ? 0 : 8;
    const unsigned vs_lds = (unsigned)(uintptr_t)&Vs[wid][0][0];

    const _Float16* qrow = Qp + (size_t)(qi0 + lm) * D_HEAD;
    const v16h qf0 = pack16(*(const v8h*)(qrow + koff), *(const v8h*)(qrow + koff + 16));
    const v16h qf1 = pack16(*(const v8h*)(qrow + 32 + koff), *(const v8h*)(qrow + 32 + koff + 16));

    v8f zero = {};
    v8f oacc[4];
#pragma unroll
    for (int ni = 0; ni < 4; ++ni) oacc[ni] = zero;
    float mrow[8], lsum[8];
#pragma unroll
    for (int r = 0; r < 8; ++r) { mrow[r] = -3.0e38f; lsum[r] = 0.0f; }

    int jlow = qi0 - (WINDOW - 1);
    if (jlow < 0) jlow = 0;
    const int jstart = (jlow >> 5) << 5;

    for (int j0 = jstart; j0 <= qi0 + 15; j0 += 32) {
        // async DMA of this wave's 32x64 V tile; overlapped with S + softmax
        tdm_load_2d(Vp + (size_t)j0 * D_HEAD, vs_lds,
                    D_HEAD, (unsigned)SEQ_L, D_HEAD, 32u, D_HEAD, 0u, 0u, 0u);

        const _Float16* krA = Kp + (size_t)(j0 + lm) * D_HEAD;
        const _Float16* krB = Kp + (size_t)(j0 + 16 + lm) * D_HEAD;
        if (j0 + 32 <= qi0 + 15)  // prefetch next key tile
            __builtin_prefetch(krA + 32 * D_HEAD, 0, 1);
        const v16h kA0 = pack16(*(const v8h*)(krA + koff), *(const v8h*)(krA + koff + 16));
        const v16h kA1 = pack16(*(const v8h*)(krA + 32 + koff), *(const v8h*)(krA + 32 + koff + 16));
        const v16h kB0 = pack16(*(const v8h*)(krB + koff), *(const v8h*)(krB + koff + 16));
        const v16h kB1 = pack16(*(const v8h*)(krB + 32 + koff), *(const v8h*)(krB + 32 + koff + 16));

        v8f s_lo = zero, s_hi = zero;
        s_lo = wmma_f16(qf0, kA0, s_lo);
        s_lo = wmma_f16(qf1, kA1, s_lo);
        s_hi = wmma_f16(qf0, kB0, s_hi);
        s_hi = wmma_f16(qf1, kB1, s_hi);

        // online softmax (reductions stay inside each 16-lane half)
        float pl[8], ph[8], alpha[8];
#pragma unroll
        for (int r = 0; r < 8; ++r) {
            const int qi = qi0 + ((lane < 16u) ? r : r + 8);
            const int jlo = j0 + lm;
            const int jhi = j0 + 16 + lm;
            float slo = s_lo[r] * 0.125f;  // dh^-0.5
            float shi = s_hi[r] * 0.125f;
            if (!((jlo <= qi) && (jlo + WINDOW > qi))) slo = -3.0e38f;
            if (!((jhi <= qi) && (jhi + WINDOW > qi))) shi = -3.0e38f;
            float mx = fmaxf(slo, shi);
#pragma unroll
            for (int msk = 1; msk < 16; msk <<= 1) mx = fmaxf(mx, __shfl_xor(mx, msk, 32));
            const float mnew = fmaxf(mrow[r], mx);
            const float a = __expf(mrow[r] - mnew);
            const float p0 = __expf(slo - mnew);
            const float p1 = __expf(shi - mnew);
            float ps = p0 + p1;
#pragma unroll
            for (int msk = 1; msk < 16; msk <<= 1) ps += __shfl_xor(ps, msk, 32);
            lsum[r] = lsum[r] * a + ps;
            mrow[r] = mnew;
            alpha[r] = a;
            pl[r] = p0;
            ph[r] = p1;
        }
#pragma unroll
        for (int ni = 0; ni < 4; ++ni)
#pragma unroll
            for (int r = 0; r < 8; ++r) oacc[ni][r] *= alpha[r];

        // bounce P through LDS: C layout -> 16-bit A layout
#pragma unroll
        for (int r = 0; r < 8; ++r) {
            const int rr = (lane < 16u) ? r : r + 8;
            Ps[wid][rr][lm] = (_Float16)pl[r];
            Ps[wid][rr][lm + 16] = (_Float16)ph[r];
        }
        asm volatile("s_wait_dscnt 0" ::: "memory");
        const _Float16* prow = &Ps[wid][lm][0];
        const v16h pf = pack16(*(const v8h*)(prow + koff), *(const v8h*)(prow + koff + 16));

        // V tile must have landed before the P*V fragments
        __builtin_amdgcn_s_wait_tensorcnt(0);
#pragma unroll
        for (int ni = 0; ni < 4; ++ni) {
            v16h vf;
#pragma unroll
            for (int t = 0; t < 8; ++t) {
                vf[t]     = Vs[wid][koff + t][ni * 16 + lm];
                vf[t + 8] = Vs[wid][koff + 16 + t][ni * 16 + lm];
            }
            oacc[ni] = wmma_f16(pf, vf, oacc[ni]);
        }
    }

    // epilogue: O row-major f16 [4096, 1024]
#pragma unroll
    for (int ni = 0; ni < 4; ++ni)
#pragma unroll
        for (int r = 0; r < 8; ++r) {
            const int qi = qi0 + ((lane < 16u) ? r : r + 8);
            const int d = ni * 16 + lm;
            const float inv = (lsum[r] > 0.0f) ? 1.0f / lsum[r] : 0.0f;
            Og[((size_t)(b * SEQ_L + qi)) * D_MODEL + hh * D_HEAD + d] =
                (_Float16)(oacc[ni][r] * inv);
        }
}

// ---------------------------------------------------------------------------
// Host-side orchestration
// ---------------------------------------------------------------------------
extern "C" void kernel_launch(void* const* d_in, const int* in_sizes, int n_in,
                              void* d_out, int out_size, void* d_ws, size_t ws_size,
                              hipStream_t stream) {
    (void)in_sizes; (void)n_in; (void)out_size; (void)ws_size;

    const float* x     = (const float*)d_in[0];
    const float* Wq    = (const float*)d_in[1];
    const float* Wk    = (const float*)d_in[2];
    const float* Wv    = (const float*)d_in[3];
    const float* Wo    = (const float*)d_in[4];
    const float* gamma = (const float*)d_in[5];
    const float* beta  = (const float*)d_in[6];

    char* ws = (char*)d_ws;
    const size_t MB = 1024 * 1024;
    _Float16* h16  = (_Float16*)(ws + 0 * MB);
    _Float16* wq16 = (_Float16*)(ws + 8 * MB);
    _Float16* wk16 = (_Float16*)(ws + 10 * MB);
    _Float16* wv16 = (_Float16*)(ws + 12 * MB);
    _Float16* wo16 = (_Float16*)(ws + 14 * MB);
    _Float16* Qb   = (_Float16*)(ws + 16 * MB);
    _Float16* Kb   = (_Float16*)(ws + 24 * MB);
    _Float16* Vb   = (_Float16*)(ws + 32 * MB);
    _Float16* O16  = (_Float16*)(ws + 40 * MB);

    const int wn = D_MODEL * D_MODEL;
    cvt_w_kernel<<<wn / 1024, 256, 0, stream>>>(Wq, wq16, wn);
    cvt_w_kernel<<<wn / 1024, 256, 0, stream>>>(Wk, wk16, wn);
    cvt_w_kernel<<<wn / 1024, 256, 0, stream>>>(Wv, wv16, wn);
    cvt_w_kernel<<<wn / 1024, 256, 0, stream>>>(Wo, wo16, wn);

    ln_kernel<<<BL, 256, 0, stream>>>(x, gamma, beta, h16);

    dim3 ggrid(D_MODEL / GT_N, BL / GT_M);  // (8, 32)
    gemm_wmma_kernel<<<ggrid, 256, 0, stream>>>(h16, wq16, Qb, nullptr,
                                                BL, D_MODEL, D_MODEL, 0);
    gemm_wmma_kernel<<<ggrid, 256, 0, stream>>>(h16, wk16, Kb, nullptr,
                                                BL, D_MODEL, D_MODEL, 0);
    gemm_wmma_kernel<<<ggrid, 256, 0, stream>>>(h16, wv16, Vb, nullptr,
                                                BL, D_MODEL, D_MODEL, 0);

    dim3 agrid(SEQ_L / (AT_WAVES * 16), BATCH * N_HEADS);  // (32, 32)
    attn_kernel<<<agrid, AT_WAVES * 32, 0, stream>>>(Qb, Kb, Vb, O16);

    gemm_wmma_kernel<<<ggrid, 256, 0, stream>>>(O16, wo16, d_out, x,
                                                BL, D_MODEL, D_MODEL, 2);
}
